// GQAManual_34686155883165
// MI455X (gfx1250) — matrix operand order repaired and checked by reading, not compiled, not run on hardware
//
#include <hip/hip_runtime.h>

// ---------------------------------------------------------------------------
// MI455X (gfx1250) GQA forward: B=1, T=4096, D=1024, HQ=16, HKV=4, HD=64.
// Compute-bound (~58 GFLOP, working set ~50 MB -> fully L2-resident at 192MB).
// All five GEMM stages ride v_wmma_f32_16x16x32_bf16 (bf16 in, f32 acc).
// Weights are pre-transposed once so GEMM operand staging is pure contiguous
// b128 traffic (via global_load_async_to_lds_b128); V is produced transposed
// so attention is barrier-free with all WMMA fragments loaded straight from
// global/L2.
// ---------------------------------------------------------------------------

typedef __attribute__((ext_vector_type(16))) __bf16 v16bf;
typedef __attribute__((ext_vector_type(8)))  float  v8f;
typedef __attribute__((ext_vector_type(4)))  unsigned int u32x4;
typedef __attribute__((__vector_size__(4 * sizeof(int)))) int i32x4v;

union Frag { v16bf bf; u32x4 u[2]; };    // 16 bf16 = 8 VGPRs (A or B operand)

__device__ __forceinline__ v8f wmma_bf16(v16bf a, v16bf b, v8f c) {
  // (neg_a, A, neg_b, B, c_mod, C, reuse_a, reuse_b)
  return __builtin_amdgcn_wmma_f32_16x16x32_bf16(false, a, false, b, (short)0, c,
                                                 false, false);
}

#if __has_builtin(__builtin_amdgcn_global_load_async_to_lds_b128)
#define ASYNC_STAGE 1
#else
#define ASYNC_STAGE 0
#endif

// ---------------------------------------------------------------------------
// f32 -> bf16 cast (row-major copy)
// ---------------------------------------------------------------------------
__global__ void castk(const float* __restrict__ in, __bf16* __restrict__ out, int n) {
  int i = (blockIdx.x * 256 + threadIdx.x) * 4;
  if (i + 3 < n) {
    float4 f = *(const float4*)(in + i);
    out[i + 0] = (__bf16)f.x;
    out[i + 1] = (__bf16)f.y;
    out[i + 2] = (__bf16)f.z;
    out[i + 3] = (__bf16)f.w;
  }
}

// f32 [K][N] -> bf16 [N][K] (one-time weight transpose; coalesced reads)
__global__ void cast_t(const float* __restrict__ in, __bf16* __restrict__ out,
                       int K, int N) {
  int i = blockIdx.x * 256 + threadIdx.x;
  if (i < K * N) {
    int k = i / N, n = i - k * N;
    out[(size_t)n * K + k] = (__bf16)in[i];
  }
}

// ---------------------------------------------------------------------------
// C[M,N] = A[M,K] @ Bt[N,K]^T, bf16 in, WMMA f32 accumulate.
// WG: 256 threads = 8 waves (2 m x 4 n), wave tile 64x32, WG tile 128x128.
// OUTMODE: 0 = bf16 row-major, 1 = f32 row-major, 2 = bf16 transposed [N][M].
// ---------------------------------------------------------------------------
template <int OUTMODE>
__global__ __launch_bounds__(256) void gemm_tt(const __bf16* __restrict__ A,
                                               const __bf16* __restrict__ Bt,
                                               void* __restrict__ Cout,
                                               int M, int N, int K) {
  constexpr int LD = 40;                       // padded stride (bf16 elems)
  __shared__ __bf16 As[128 * LD];              // A  tile 128(m) x 32(k)
  __shared__ __bf16 Bs[128 * LD];              // Bt tile 128(n) x 32(k)

  const int tid  = threadIdx.x;
  const int lane = tid & 31;
  const int w    = tid >> 5;
  const int lh   = lane >> 4;                  // lane half
  const int mr   = lane & 15;
  const int mw   = (w & 1) * 64;               // wave m-origin in WG tile
  const int nw   = (w >> 1) * 32;              // wave n-origin in WG tile
  const int tileM = blockIdx.y * 128;
  const int tileN = blockIdx.x * 128;

  v8f acc[4][2];
  #pragma unroll
  for (int mi = 0; mi < 4; ++mi)
    #pragma unroll
    for (int ni = 0; ni < 2; ++ni) acc[mi][ni] = {};

  const int srow = tid >> 1, scol = (tid & 1) * 16;  // 128 rows x 32 cols

  for (int kb = 0; kb < K; kb += 32) {
    const __bf16* ga = A  + (size_t)(tileM + srow) * K + kb + scol;
    const __bf16* gb = Bt + (size_t)(tileN + srow) * K + kb + scol;
    __bf16* la = &As[srow * LD + scol];
    __bf16* lb = &Bs[srow * LD + scol];
#if ASYNC_STAGE
    __builtin_amdgcn_global_load_async_to_lds_b128((i32x4v*)ga, (i32x4v*)la, 0, 0);
    __builtin_amdgcn_global_load_async_to_lds_b128((i32x4v*)(ga + 8),
                                                   (i32x4v*)(la + 8), 0, 0);
    __builtin_amdgcn_global_load_async_to_lds_b128((i32x4v*)gb, (i32x4v*)lb, 0, 0);
    __builtin_amdgcn_global_load_async_to_lds_b128((i32x4v*)(gb + 8),
                                                   (i32x4v*)(lb + 8), 0, 0);
    asm volatile("s_wait_asynccnt 0" ::: "memory");
#else
    *(u32x4*)(la)     = *(const u32x4*)(ga);
    *(u32x4*)(la + 8) = *(const u32x4*)(ga + 8);
    *(u32x4*)(lb)     = *(const u32x4*)(gb);
    *(u32x4*)(lb + 8) = *(const u32x4*)(gb + 8);
#endif
    __syncthreads();

    // A fragment: lane 16*lh+m holds K = 16*lh .. 16*lh+15
    Frag a[4], b[2];
    #pragma unroll
    for (int mi = 0; mi < 4; ++mi) {
      const __bf16* p = &As[(mw + mi * 16 + mr) * LD];
      a[mi].u[0] = *(const u32x4*)(p + 16 * lh);
      a[mi].u[1] = *(const u32x4*)(p + 16 * lh + 8);
    }
    // B fragment: lane 16*lh+n holds K = 16*lh .. 16*lh+15 (rows of Bt)
    #pragma unroll
    for (int ni = 0; ni < 2; ++ni) {
      const __bf16* p = &Bs[(nw + ni * 16 + mr) * LD + 16 * lh];
      b[ni].u[0] = *(const u32x4*)(p);
      b[ni].u[1] = *(const u32x4*)(p + 8);
    }
    #pragma unroll
    for (int mi = 0; mi < 4; ++mi)
      #pragma unroll
      for (int ni = 0; ni < 2; ++ni)
        acc[mi][ni] = wmma_bf16(a[mi].bf, b[ni].bf, acc[mi][ni]);
    __syncthreads();
  }

  // C/D layout: VGPR r, lanes 0-15 -> (M=r, N=lane); lanes 16-31 -> (M=8+r)
  #pragma unroll
  for (int mi = 0; mi < 4; ++mi)
    #pragma unroll
    for (int ni = 0; ni < 2; ++ni)
      #pragma unroll
      for (int r = 0; r < 8; ++r) {
        int row = tileM + mw + mi * 16 + r + 8 * lh;
        int col = tileN + nw + ni * 16 + mr;
        if (OUTMODE == 1)
          ((float*)Cout)[(size_t)row * N + col] = acc[mi][ni][r];
        else if (OUTMODE == 2)
          ((__bf16*)Cout)[(size_t)col * M + row] = (__bf16)acc[mi][ni][r];
        else
          ((__bf16*)Cout)[(size_t)row * N + col] = (__bf16)acc[mi][ni][r];
      }
}

// ---------------------------------------------------------------------------
// Causal GQA flash attention, barrier-free. Grid: (T/64, HQ), 128 threads =
// 4 independent waves, wave w owns 16 q rows. Q [T,1024], K [T,256] and
// Vt [256,T] all give contiguous-b128 WMMA fragments straight from global
// (K/V tiles are L2-resident). LDS only for the per-wave P C->A relayout.
// Per 64-key tile: 8 WMMA for S=QK^T, 8 WMMA for O+=PV.
// ---------------------------------------------------------------------------
__global__ __launch_bounds__(128) void attn_fwd(const __bf16* __restrict__ Qb,
                                                const __bf16* __restrict__ Kb,
                                                const __bf16* __restrict__ Vt,
                                                __bf16* __restrict__ Cb) {
  constexpr int LDP = 72;
  __shared__ __bf16 Ps[4 * 16 * LDP];          // per-wave P staging

  const int tid  = threadIdx.x;
  const int lane = tid & 31;
  const int w    = tid >> 5;
  const int lh   = lane >> 4;
  const int mr   = lane & 15;
  const int qt   = blockIdx.x;                 // 64-row q tile
  const int qh   = blockIdx.y;                 // q head (= h*G + g)
  const int h    = qh >> 2;                    // kv head (G = 4)
  const int qrow = qt * 64 + w * 16;           // wave's first q row

  // Q fragments for d = 0..63 (two K=32 steps), loaded once.
  Frag qa[2];
  {
    const __bf16* qp = Qb + (size_t)(qrow + mr) * 1024 + qh * 64;
    #pragma unroll
    for (int s = 0; s < 2; ++s) {
      qa[s].u[0] = *(const u32x4*)(qp + s * 32 + 16 * lh);
      qa[s].u[1] = *(const u32x4*)(qp + s * 32 + 16 * lh + 8);
    }
  }

  v8f o[4];
  #pragma unroll
  for (int dt = 0; dt < 4; ++dt) o[dt] = {};
  float mrow[8], lrow[8];
  #pragma unroll
  for (int r = 0; r < 8; ++r) { mrow[r] = -3.0e38f; lrow[r] = 0.f; }

  __bf16* pw = &Ps[w * 16 * LDP];

  for (int kt = 0; kt <= qt; ++kt) {
    const int k0 = kt * 64;

    // S = Q K^T * 1/sqrt(64); B fragments are rows of K (contiguous in d)
    v8f s[4];
    #pragma unroll
    for (int nt = 0; nt < 4; ++nt) {
      v8f a = {};
      #pragma unroll
      for (int ds = 0; ds < 2; ++ds) {
        Frag b;
        const __bf16* kp =
            Kb + (size_t)(k0 + nt * 16 + mr) * 256 + h * 64 + ds * 32 + 16 * lh;
        b.u[0] = *(const u32x4*)(kp);
        b.u[1] = *(const u32x4*)(kp + 8);
        a = wmma_bf16(qa[ds].bf, b.bf, a);
      }
      s[nt] = a * 0.125f;
    }

    if (kt == qt) {  // diagonal tile: causal mask
      #pragma unroll
      for (int nt = 0; nt < 4; ++nt)
        #pragma unroll
        for (int r = 0; r < 8; ++r) {
          int qg = qrow + r + 8 * lh;
          int kg = k0 + nt * 16 + mr;
          if (kg > qg) s[nt][r] = -1.0e30f;
        }
    }

    // Online softmax: row M=r+8*lh lives across the lane's 16-lane half.
    #pragma unroll
    for (int r = 0; r < 8; ++r) {
      float tm = fmaxf(fmaxf(s[0][r], s[1][r]), fmaxf(s[2][r], s[3][r]));
      #pragma unroll
      for (int off = 1; off < 16; off <<= 1)
        tm = fmaxf(tm, __shfl_xor(tm, off, 32));
      float nm    = fmaxf(mrow[r], tm);
      float alpha = __expf(mrow[r] - nm);
      mrow[r] = nm;
      float rs = 0.f;
      #pragma unroll
      for (int nt = 0; nt < 4; ++nt) {
        float p = __expf(s[nt][r] - nm);
        s[nt][r] = p;
        rs += p;
      }
      #pragma unroll
      for (int off = 1; off < 16; off <<= 1)
        rs += __shfl_xor(rs, off, 32);
      lrow[r] = lrow[r] * alpha + rs;
      #pragma unroll
      for (int dt = 0; dt < 4; ++dt) o[dt][r] *= alpha;
    }

    // P: C-layout regs -> LDS (bf16) -> reload in A-layout (same wave only,
    // DS ops from one wave are in-order; just wait out the DS counter).
    #pragma unroll
    for (int nt = 0; nt < 4; ++nt)
      #pragma unroll
      for (int r = 0; r < 8; ++r)
        pw[(r + 8 * lh) * LDP + nt * 16 + mr] = (__bf16)s[nt][r];
    asm volatile("s_wait_dscnt 0" ::: "memory");

    // O += P V  (B fragments are rows of Vt = columns of V, contiguous in t)
    #pragma unroll
    for (int ks = 0; ks < 2; ++ks) {
      Frag a;
      const __bf16* pp = &pw[mr * LDP + ks * 32];
      a.u[0] = *(const u32x4*)(pp + 16 * lh);
      a.u[1] = *(const u32x4*)(pp + 16 * lh + 8);
      #pragma unroll
      for (int dt = 0; dt < 4; ++dt) {
        Frag b;
        const __bf16* vp =
            Vt + (size_t)(h * 64 + dt * 16 + mr) * 4096 + k0 + ks * 32 + 16 * lh;
        b.u[0] = *(const u32x4*)(vp);
        b.u[1] = *(const u32x4*)(vp + 8);
        o[dt] = wmma_bf16(a.bf, b.bf, o[dt]);
      }
    }
  }

  // ctx[row][qh*64 + d] = O / l   (bf16 for the final WMMA GEMM)
  #pragma unroll
  for (int dt = 0; dt < 4; ++dt)
    #pragma unroll
    for (int r = 0; r < 8; ++r) {
      float val = o[dt][r] / lrow[r];
      Cb[(size_t)(qrow + r + 8 * lh) * 1024 + qh * 64 + dt * 16 + mr] =
          (__bf16)val;
    }
}

// ---------------------------------------------------------------------------
// Launch
// ---------------------------------------------------------------------------
extern "C" void kernel_launch(void* const* d_in, const int* in_sizes, int n_in,
                              void* d_out, int out_size, void* d_ws, size_t ws_size,
                              hipStream_t stream) {
  const float* x  = (const float*)d_in[0];   // [4096,1024]
  const float* Wq = (const float*)d_in[1];   // [1024,1024]
  const float* Wk = (const float*)d_in[2];   // [1024, 256]
  const float* Wv = (const float*)d_in[3];   // [1024, 256]
  const float* Wo = (const float*)d_in[4];   // [1024,1024]
  float* out = (float*)d_out;                // [4096,1024]

  char* ws = (char*)d_ws;
  __bf16* xb   = (__bf16*)(ws);                       // x    [4096,1024] 8 MiB
  __bf16* Wqt  = (__bf16*)(ws + (8u << 20));          // Wq^T [1024,1024] 2 MiB
  __bf16* Wkt  = (__bf16*)(ws + (10u << 20));         // Wk^T [ 256,1024] 512 KiB
  __bf16* Wvt  = (__bf16*)(ws + (10u << 20) + (512u << 10));
  __bf16* Wot  = (__bf16*)(ws + (11u << 20));         // Wo^T [1024,1024] 2 MiB
  __bf16* Qb   = (__bf16*)(ws + (13u << 20));         // Q    [4096,1024] 8 MiB
  __bf16* Kb   = (__bf16*)(ws + (21u << 20));         // K    [4096, 256] 2 MiB
  __bf16* Vtb  = (__bf16*)(ws + (23u << 20));         // V^T  [ 256,4096] 2 MiB
  __bf16* Cb   = (__bf16*)(ws + (25u << 20));         // ctx  [4096,1024] 8 MiB

  const int T = 4096, D = 1024, DKV = 256;

  castk<<<(T * D) / 1024, 256, 0, stream>>>(x, xb, T * D);
  cast_t<<<(D * D) / 256, 256, 0, stream>>>(Wq, Wqt, D, D);
  cast_t<<<(D * DKV) / 256, 256, 0, stream>>>(Wk, Wkt, D, DKV);
  cast_t<<<(D * DKV) / 256, 256, 0, stream>>>(Wv, Wvt, D, DKV);
  cast_t<<<(D * D) / 256, 256, 0, stream>>>(Wo, Wot, D, D);

  // Projections (A [T,K] @ Wt [N,K]^T): Q, K row-major; V written transposed.
  gemm_tt<0><<<dim3(D / 128, T / 128), 256, 0, stream>>>(xb, Wqt, Qb, T, D, D);
  gemm_tt<0><<<dim3(DKV / 128, T / 128), 256, 0, stream>>>(xb, Wkt, Kb, T, DKV, D);
  gemm_tt<2><<<dim3(DKV / 128, T / 128), 256, 0, stream>>>(xb, Wvt, Vtb, T, DKV, D);

  // Causal GQA flash attention -> ctx (bf16)
  attn_fwd<<<dim3(T / 64, 16), 128, 0, stream>>>(Qb, Kb, Vtb, Cb);

  // Output projection: out = ctx @ Wo (f32 out)
  gemm_tt<1><<<dim3(D / 128, T / 128), 256, 0, stream>>>(Cb, Wot, out, T, D, D);
}
